// GraphEncoder_65661460021322
// MI455X (gfx1250) — matrix operand order repaired and checked by reading, not compiled, not run on hardware
//
#include <hip/hip_runtime.h>
#include <stdint.h>

// ---------------------------------------------------------------------------
// CDNA5 (gfx1250) GAT encoder.
//  - GEMMs: bf16 WMMA (v_wmma_f32_16x16x32_bf16), wave32, 8 waves / 128x64 tile,
//    double-buffered LDS, all staging via b128 (weights pre-transposed bf16).
//  - Activations kept bf16 between layers (fused into BN+ReLU epilogue) so the
//    A matrix (~100 MB) stays L2-resident on the 192 MB L2.
//  - Edge softmax/aggregation: ordered-uint atomicMax + float atomics (HBM/atomic
//    bound; this is the roofline-dominant part at ~1.8 GB of gathers per layer).
// ---------------------------------------------------------------------------

typedef __attribute__((ext_vector_type(16))) __bf16 v16bf;
typedef __attribute__((ext_vector_type(8)))  __bf16 bf16x8;
typedef __attribute__((ext_vector_type(8)))  float  v8f;

static constexpr int   kN       = 50000;
static constexpr int   kE       = 400000;
static constexpr int   kEtot    = kE + kN;       // + self loops
static constexpr int   kNodeDim = 384;
static constexpr int   kHid     = 256;
static constexpr int   kOut     = 128;
static constexpr int   kHeads   = 4;
static constexpr int   kD0      = kHid * kHeads; // 1024
static constexpr float kNegSlope = 0.2f;
static constexpr float kBnEps    = 1e-5f;
static constexpr unsigned kEncNegInf = 0x007fffffu;  // enc_ord(-inf)

// ----- GEMM tiling ---------------------------------------------------------
#define BM 128
#define BN 64
#define BK 32
#define LDA (BK + 8)   // bf16 elems; pad to dodge LDS bank conflicts
#define LDB (BK + 8)

// monotone float <-> uint mapping so atomicMax(uint) == float max
__device__ __forceinline__ unsigned enc_ord(float f) {
  int i = __float_as_int(f);
  return (i >= 0) ? ((unsigned)i | 0x80000000u) : ~(unsigned)i;
}
__device__ __forceinline__ float dec_ord(unsigned u) {
  int i = (u & 0x80000000u) ? (int)(u & 0x7fffffffu) : (int)~u;
  return __int_as_float(i);
}

// ---------------------------------------------------------------------------
__global__ void k_cvt_bf16(const float* __restrict__ in, __bf16* __restrict__ out, int n) {
  int i = blockIdx.x * blockDim.x + threadIdx.x;
  if (i < n) out[i] = (__bf16)in[i];
}

// W[K,Nc] f32 -> Wt[Nc,K] bf16 (so GEMM B staging is contiguous b128)
__global__ void k_cvt_w_bf16_t(const float* __restrict__ in, __bf16* __restrict__ out,
                               int K, int Nc) {
  int i = blockIdx.x * blockDim.x + threadIdx.x;
  if (i >= K * Nc) return;
  int k = i / Nc, n = i - k * Nc;
  out[(size_t)n * K + k] = (__bf16)in[i];
}

__global__ void k_fill_u32(unsigned* __restrict__ p, unsigned v, int n) {
  int i = blockIdx.x * blockDim.x + threadIdx.x;
  if (i < n) p[i] = v;
}

// ---------------------------------------------------------------------------
// staging helpers: pure 16B copies into LDS
__device__ __forceinline__ void stage_a(__bf16 (*As)[LDA], const __bf16* __restrict__ A,
                                        int tid, int block_m, int k0, int M, int K) {
  const int ar = tid >> 1;            // 0..127
  const int ac = (tid & 1) << 4;      // 0,16
  const int gr = block_m + ar;
  uint4 u0 = {0, 0, 0, 0}, u1 = {0, 0, 0, 0};
  if (gr < M) {
    const uint4* src = (const uint4*)(A + (size_t)gr * K + k0 + ac);
    u0 = src[0];
    u1 = src[1];
    if (k0 + BK < K) __builtin_prefetch(A + (size_t)gr * K + k0 + BK + ac, 0, 1);
  }
  *(uint4*)&As[ar][ac]     = u0;
  *(uint4*)&As[ar][ac + 8] = u1;
}

__device__ __forceinline__ void stage_b(__bf16 (*Bs)[LDB], const __bf16* __restrict__ Bt,
                                        int tid, int block_n, int k0, int K) {
  const int br = tid >> 2;            // 0..63
  const int bc = (tid & 3) << 3;      // 0,8,16,24
  const uint4* src = (const uint4*)(Bt + (size_t)(block_n + br) * K + k0 + bc);
  *(uint4*)&Bs[br][bc] = src[0];
}

// C[M,Nc] = A[M,K](bf16, row-major) x Bt[Nc,K](bf16, pre-transposed)
__global__ __launch_bounds__(256)
void k_gemm_bf16_wmma(const __bf16* __restrict__ A, const __bf16* __restrict__ Bt,
                      float* __restrict__ C, int M, int K, int Nc) {
  __shared__ __bf16 As[2][BM][LDA];
  __shared__ __bf16 Bs[2][BN][LDB];

  const int tid  = threadIdx.x;
  const int lane = tid & 31;
  const int wave = tid >> 5;
  const int wm   = wave & 1;       // 0..1 -> 64 rows
  const int wn   = wave >> 1;      // 0..3 -> 16 cols
  const int half = lane >> 4;      // 0..1
  const int l16  = lane & 15;

  const int block_m = blockIdx.y * BM;
  const int block_n = blockIdx.x * BN;

  const v8f vzero = {0.f, 0.f, 0.f, 0.f, 0.f, 0.f, 0.f, 0.f};
  v8f acc[4];
#pragma unroll
  for (int s = 0; s < 4; ++s) acc[s] = vzero;

  // prologue: stage tile 0 into buffer 0
  stage_a(As[0], A, tid, block_m, 0, M, K);
  stage_b(Bs[0], Bt, tid, block_n, 0, K);
  __syncthreads();

  const int nk = K / BK;
  for (int kt = 0; kt < nk; ++kt) {
    const int cur = kt & 1;
    if (kt + 1 < nk) {  // stage next tile into the other buffer (overlaps WMMAs)
      stage_a(As[cur ^ 1], A, tid, block_m, (kt + 1) * BK, M, K);
      stage_b(Bs[cur ^ 1], Bt, tid, block_n, (kt + 1) * BK, K);
    }
    // B fragment: lane holds col N=l16; K = half*16 + 0..15 (contiguous)
    v16bf bfrag;
    {
      const __bf16* bp = &Bs[cur][wn * 16 + l16][half * 16];
      bf16x8 b0 = *(const bf16x8*)bp;
      bf16x8 b1 = *(const bf16x8*)(bp + 8);
      bfrag = __builtin_shufflevector(b0, b1, 0,1,2,3,4,5,6,7,8,9,10,11,12,13,14,15);
    }
#pragma unroll
    for (int s = 0; s < 4; ++s) {
      const int r = wm * 64 + s * 16 + l16;
      // lanes 0-15: K 0..15 ; lanes 16-31: K 16..31 (contiguous)
      bf16x8 a0 = *(const bf16x8*)&As[cur][r][half * 16];
      bf16x8 a1 = *(const bf16x8*)&As[cur][r][half * 16 + 8];
      v16bf afrag = __builtin_shufflevector(a0, a1, 0,1,2,3,4,5,6,7,8,9,10,11,12,13,14,15);
      acc[s] = __builtin_amdgcn_wmma_f32_16x16x32_bf16(
          false, afrag, false, bfrag, (short)0, acc[s], false, false);
    }
    __syncthreads();
  }

  // epilogue: VGPR j -> row (half*8 + j), col l16
  const int col = block_n + wn * 16 + l16;
#pragma unroll
  for (int s = 0; s < 4; ++s) {
    const int row0 = block_m + wm * 64 + s * 16 + half * 8;
#pragma unroll
    for (int j = 0; j < 8; ++j) {
      const int row = row0 + j;
      if (row < M) C[(size_t)row * Nc + col] = acc[s][j];
    }
  }
}

// ---------------------------------------------------------------------------
// es[n,h] = <h[n,h,:], a_s[h,:]> ; ed likewise
__global__ void k_attn_scores(const float* __restrict__ h,
                              const float* __restrict__ a_s,
                              const float* __restrict__ a_d,
                              float* __restrict__ es, float* __restrict__ ed,
                              int H, int C) {
  int i = blockIdx.x * blockDim.x + threadIdx.x;
  if (i >= kN * H) return;
  int n = i / H, hd = i - n * H;
  const float4* hp = (const float4*)(h + ((size_t)n * H + hd) * C);
  const float4* sp = (const float4*)(a_s + (size_t)hd * C);
  const float4* dp = (const float4*)(a_d + (size_t)hd * C);
  float s = 0.f, d = 0.f;
  for (int c = 0; c < C / 4; ++c) {
    float4 hv = hp[c], sv = sp[c], dv = dp[c];
    s += hv.x * sv.x + hv.y * sv.y + hv.z * sv.z + hv.w * sv.w;
    d += hv.x * dv.x + hv.y * dv.y + hv.z * dv.z + hv.w * dv.w;
  }
  es[i] = s; ed[i] = d;
}

// leaky_relu(es[src]+ed[dst]); segment max over dst via ordered atomicMax
__global__ void k_edge_lrelu_max(const int* __restrict__ ei,
                                 const float* __restrict__ es, const float* __restrict__ ed,
                                 float* __restrict__ ev, unsigned* __restrict__ menc, int H) {
  int i = blockIdx.x * blockDim.x + threadIdx.x;
  if (i >= kEtot * H) return;
  int e = i / H, hd = i - e * H;
  int s = (e < kE) ? ei[e] : (e - kE);
  int d = (e < kE) ? ei[kE + e] : (e - kE);
  float v = es[s * H + hd] + ed[d * H + hd];
  v = (v > 0.f) ? v : v * kNegSlope;
  ev[i] = v;
  atomicMax(&menc[d * H + hd], enc_ord(v));
}

// p = exp(v - max[dst]); segment sum over dst
__global__ void k_edge_exp_sum(const int* __restrict__ ei, float* __restrict__ ev,
                               const unsigned* __restrict__ menc,
                               float* __restrict__ denom, int H) {
  int i = blockIdx.x * blockDim.x + threadIdx.x;
  if (i >= kEtot * H) return;
  int e = i / H, hd = i - e * H;
  int d = (e < kE) ? ei[kE + e] : (e - kE);
  float p = __expf(ev[i] - dec_ord(menc[d * H + hd]));
  ev[i] = p;
  atomicAdd(&denom[d * H + hd], p);
}

// agg[dst,h,:] += h[src,h,:] * (p / denom[dst,h]) ; 4 channels per thread
__global__ void k_edge_aggregate(const int* __restrict__ ei, const float* __restrict__ h,
                                 const float* __restrict__ ev, const float* __restrict__ denom,
                                 float* __restrict__ agg, int H, int C) {
  const int chunks = C >> 2;
  int i = blockIdx.x * blockDim.x + threadIdx.x;
  if (i >= kEtot * H * chunks) return;
  int c4 = i % chunks;
  int eh = i / chunks;
  int e  = eh / H, hd = eh - e * H;
  int s = (e < kE) ? ei[e] : (e - kE);
  int d = (e < kE) ? ei[kE + e] : (e - kE);
  float alpha = ev[eh] / denom[d * H + hd];
  float4 hv = *(const float4*)(h + ((size_t)s * H + hd) * C + c4 * 4);
  float* o = agg + ((size_t)d * H + hd) * C + c4 * 4;
  atomicAdd(o + 0, hv.x * alpha);
  atomicAdd(o + 1, hv.y * alpha);
  atomicAdd(o + 2, hv.z * alpha);
  atomicAdd(o + 3, hv.w * alpha);
}

// relu(batchnorm(agg + bias)) -> bf16 activation for the next GEMM
__global__ void k_bias_bn_relu_bf16(const float* __restrict__ agg, const float* __restrict__ b,
                                    const float* __restrict__ g,  const float* __restrict__ bt,
                                    const float* __restrict__ mn, const float* __restrict__ vr,
                                    __bf16* __restrict__ out, int D) {
  int i = blockIdx.x * blockDim.x + threadIdx.x;
  if (i >= kN * D) return;
  int c = i % D;
  float v = agg[i] + b[c];
  v = (v - mn[c]) * __frsqrt_rn(vr[c] + kBnEps) * g[c] + bt[c];
  out[i] = (__bf16)(v > 0.f ? v : 0.f);
}

__global__ void k_add_bias(const float* __restrict__ agg, const float* __restrict__ b,
                           float* __restrict__ out, int D) {
  int i = blockIdx.x * blockDim.x + threadIdx.x;
  if (i >= kN * D) return;
  out[i] = agg[i] + b[i % D];
}

// ---------------------------------------------------------------------------
static inline int cdiv(int a, int b) { return (a + b - 1) / b; }

extern "C" void kernel_launch(void* const* d_in, const int* in_sizes, int n_in,
                              void* d_out, int out_size, void* d_ws, size_t ws_size,
                              hipStream_t stream) {
  (void)in_sizes; (void)n_in; (void)out_size; (void)ws_size;
  const float* x   = (const float*)d_in[0];
  const int*   ei  = (const int*)d_in[1];
  const float* W[3]    = {(const float*)d_in[2],  (const float*)d_in[10], (const float*)d_in[18]};
  const float* a_s[3]  = {(const float*)d_in[3],  (const float*)d_in[11], (const float*)d_in[19]};
  const float* a_d[3]  = {(const float*)d_in[4],  (const float*)d_in[12], (const float*)d_in[20]};
  const float* bia[3]  = {(const float*)d_in[5],  (const float*)d_in[13], (const float*)d_in[21]};
  const float* bng[2]  = {(const float*)d_in[6],  (const float*)d_in[14]};
  const float* bnb[2]  = {(const float*)d_in[7],  (const float*)d_in[15]};
  const float* bnm[2]  = {(const float*)d_in[8],  (const float*)d_in[16]};
  const float* bnv[2]  = {(const float*)d_in[9],  (const float*)d_in[17]};

  // workspace carve-up (~525 MB)
  char* w = (char*)d_ws;
  float*    bufA  = (float*)w;    w += (size_t)kN * kD0 * sizeof(float);    // gemm out
  float*    bufB  = (float*)w;    w += (size_t)kN * kD0 * sizeof(float);    // agg target
  __bf16*   actbf = (__bf16*)w;   w += (size_t)kN * kD0 * sizeof(__bf16);   // bf16 activations
  __bf16*   Wt    = (__bf16*)w;   w += (size_t)kD0 * kD0 * sizeof(__bf16);  // bf16 W^T
  float*    es    = (float*)w;    w += (size_t)kN * kHeads * sizeof(float);
  float*    ed    = (float*)w;    w += (size_t)kN * kHeads * sizeof(float);
  unsigned* menc  = (unsigned*)w; w += (size_t)kN * kHeads * sizeof(unsigned);
  float*    denom = (float*)w;    w += (size_t)kN * kHeads * sizeof(float);
  float*    ev    = (float*)w;    w += (size_t)kEtot * kHeads * sizeof(float);

  const int Kin[3] = {kNodeDim, kD0, kD0};
  const int Hh[3]  = {kHeads, kHeads, 1};
  const int Cc[3]  = {kHid, kHid, kOut};

  // x -> bf16 once
  k_cvt_bf16<<<cdiv(kN * kNodeDim, 256), 256, 0, stream>>>(x, actbf, kN * kNodeDim);

  for (int l = 0; l < 3; ++l) {
    const int D = Hh[l] * Cc[l];
    // W -> bf16, transposed [D, Kin]
    k_cvt_w_bf16_t<<<cdiv(Kin[l] * D, 256), 256, 0, stream>>>(W[l], Wt, Kin[l], D);
    // h = act @ W (WMMA, bf16 in / f32 out)
    k_gemm_bf16_wmma<<<dim3(D / BN, cdiv(kN, BM)), 256, 0, stream>>>(
        actbf, Wt, bufA, kN, Kin[l], D);
    // attention scores per node/head
    k_attn_scores<<<cdiv(kN * Hh[l], 256), 256, 0, stream>>>(
        bufA, a_s[l], a_d[l], es, ed, Hh[l], Cc[l]);
    // init segment-max / denom / aggregation target
    k_fill_u32<<<cdiv(kN * Hh[l], 256), 256, 0, stream>>>(menc, kEncNegInf, kN * Hh[l]);
    k_fill_u32<<<cdiv(kN * Hh[l], 256), 256, 0, stream>>>((unsigned*)denom, 0u, kN * Hh[l]);
    k_fill_u32<<<cdiv(kN * D, 256), 256, 0, stream>>>((unsigned*)bufB, 0u, kN * D);
    // edge softmax
    k_edge_lrelu_max<<<cdiv(kEtot * Hh[l], 256), 256, 0, stream>>>(ei, es, ed, ev, menc, Hh[l]);
    k_edge_exp_sum<<<cdiv(kEtot * Hh[l], 256), 256, 0, stream>>>(ei, ev, menc, denom, Hh[l]);
    // weighted aggregation
    k_edge_aggregate<<<cdiv(kEtot * Hh[l] * (Cc[l] / 4), 256), 256, 0, stream>>>(
        ei, bufA, ev, denom, bufB, Hh[l], Cc[l]);
    // epilogue
    if (l < 2) {
      k_bias_bn_relu_bf16<<<cdiv(kN * D, 256), 256, 0, stream>>>(
          bufB, bia[l], bng[l], bnb[l], bnm[l], bnv[l], actbf, D);
    } else {
      k_add_bias<<<cdiv(kN * D, 256), 256, 0, stream>>>(bufB, bia[l], (float*)d_out, D);
    }
  }
}
